// Attention_26439818674221
// MI455X (gfx1250) — matrix operand order repaired and checked by reading, not compiled, not run on hardware
//
#include <hip/hip_runtime.h>
#include <hip/hip_bf16.h>
#include <math.h>

typedef __attribute__((ext_vector_type(16))) _Float16 v16h;
typedef __attribute__((ext_vector_type(8)))  _Float16 v8h;
typedef __attribute__((ext_vector_type(8)))  float    v8f;

union F16x16 { v16h v; _Float16 h[16]; unsigned int u[8]; };

#define B_    2
#define SEQ   4096
#define DIM   1024
#define HEADS 8
#define DHEAD 64
#define DHID  512          // HEADS*DHEAD
#define DH3   1536         // 3*DHID
#define ROWS  (B_*SEQ)     // 8192

// ---------------------------------------------------------------------------
// CDNA5 async global->LDS copy (ASYNCcnt-tracked, bypasses VGPRs).
// VDST = LDS byte address VGPR, VADDR = 64-bit global address.
// Generic __shared__ pointers carry the LDS offset in their low 32 bits.
// ---------------------------------------------------------------------------
__device__ __forceinline__ void async_copy_b128(const _Float16* g,
                                                _Float16* l) {
  unsigned lds = (unsigned)(unsigned long long)(void*)l;
  unsigned long long ga = (unsigned long long)(const void*)g;
  asm volatile("global_load_async_to_lds_b128 %0, %1, off"
               :: "v"(lds), "v"(ga) : "memory");
}

// ---------------------------------------------------------------------------
// 1) RMSNorm: x fp32 [8192,1024] -> xn fp16
// ---------------------------------------------------------------------------
__global__ __launch_bounds__(256) void rmsnorm_kernel(
    const float* __restrict__ x, const float* __restrict__ w,
    _Float16* __restrict__ xn) {
  const int row = blockIdx.x;
  const int tid = threadIdx.x;
  const float* xr = x + (size_t)row * DIM;
  float ss = 0.f;
  for (int j = tid; j < DIM; j += 256) { float v = xr[j]; ss += v * v; }
  #pragma unroll
  for (int m = 16; m >= 1; m >>= 1) ss += __shfl_xor(ss, m, 32);
  __shared__ float red[8];
  const int wid = tid >> 5, lane = tid & 31;
  if (lane == 0) red[wid] = ss;
  __syncthreads();
  float tot = 0.f;
  #pragma unroll
  for (int i = 0; i < 8; i++) tot += red[i];
  const float scale = rsqrtf(tot * (1.0f / DIM) + 1.1920929e-07f);
  _Float16* xo = xn + (size_t)row * DIM;
  for (int j = tid; j < DIM; j += 256)
    xo[j] = (_Float16)(xr[j] * scale * w[j]);
}

// ---------------------------------------------------------------------------
// 2) fp32 -> fp16 conversion (weights)
// ---------------------------------------------------------------------------
__global__ void f32_to_f16_kernel(const float* __restrict__ in,
                                  _Float16* __restrict__ out, int n) {
  int i = blockIdx.x * 256 + threadIdx.x;
  if (i < n) out[i] = (_Float16)in[i];
}

// ---------------------------------------------------------------------------
// 3) WMMA GEMM: C[M,N] = A[M,K]*B[K,N], fp16 in, f32 accum.
//    Block tile 128x256, BK=32, 256 threads = 8 waves (2 M x 4 N),
//    wave tile 64x64 = 16 WMMAs / K-step.
//    Double-buffered LDS staged with global_load_async_to_lds_b128;
//    counted s_wait_asynccnt overlaps next-tile copy with current compute.
// ---------------------------------------------------------------------------
template <bool OUT_F32>
__global__ __launch_bounds__(256) void gemm_f16_kernel(
    const _Float16* __restrict__ A, const _Float16* __restrict__ Bm,
    void* __restrict__ Cout, int M, int N, int K) {
  __shared__ _Float16 sA[2][128][48];    // [buf][m][k], 96B rows (16B aligned)
  __shared__ _Float16 sB[2][32][264];    // [buf][k][n], 528B rows (16B aligned)
  const int tid  = threadIdx.x;
  const int lane = tid & 31, wid = tid >> 5;
  const int gm = blockIdx.y * 128, gn = blockIdx.x * 256;
  const int wm = (wid & 1) * 64;         // wave M offset
  const int wn = (wid >> 1) * 64;        // wave N offset
  const int lm = lane & 15, hi = lane >> 4;

  v8f acc[4][4];
  #pragma unroll
  for (int i = 0; i < 4; i++)
    #pragma unroll
    for (int j = 0; j < 4; j++) acc[i][j] = (v8f)(0.f);

  // async-stage one 128x32 A tile + 32x256 B tile: 6 b128 ops per thread
  auto stage = [&](int buf, int k0) {
    #pragma unroll
    for (int i = 0; i < 2; i++) {        // A: 512 chunks
      int c = tid + i * 256;
      int r = c >> 2, c8 = (c & 3) * 8;
      async_copy_b128(&A[(size_t)(gm + r) * K + k0 + c8], &sA[buf][r][c8]);
    }
    #pragma unroll
    for (int i = 0; i < 4; i++) {        // B: 1024 chunks
      int c = tid + i * 256;
      int r = c >> 5, c8 = (c & 31) * 8;
      async_copy_b128(&Bm[(size_t)(k0 + r) * N + gn + c8], &sB[buf][r][c8]);
    }
  };

  stage(0, 0);
  const int nk = K / 32;
  for (int it = 0; it < nk; it++) {
    const int buf = it & 1;
    __syncthreads();                     // all readers of buf^1 are done
    if (it + 1 < nk) {
      stage(buf ^ 1, (it + 1) * 32);     // prefetch next tile (6 more in flight)
      asm volatile("s_wait_asynccnt 6" ::: "memory");  // drain current tile only
    } else {
      asm volatile("s_wait_asynccnt 0" ::: "memory");
    }
    __syncthreads();                     // buf visible to all waves

    // A fragments: lane holds row M=lm; K split per ISA 16-bit A layout
    F16x16 afr[4];
    #pragma unroll
    for (int s = 0; s < 4; s++) {
      int m = wm + s * 16 + lm;
      #pragma unroll
      for (int v = 0; v < 8; v++) {
        int kk = ((v < 4) ? 0 : 16) + hi * 8 + (v & 3) * 2;
        afr[s].u[v] = *(const unsigned int*)&sA[buf][m][kk];
      }
    }
    // B fragments: lane holds col N=lm; K split per ISA 16-bit layout (same as A)
    #pragma unroll
    for (int t = 0; t < 4; t++) {
      F16x16 bf;
      #pragma unroll
      for (int e = 0; e < 16; e++)
        bf.h[e] = sB[buf][((e < 8) ? 0 : 16) + hi * 8 + (e & 7)][wn + t * 16 + lm];
      #pragma unroll
      for (int s = 0; s < 4; s++)
        acc[s][t] = __builtin_amdgcn_wmma_f32_16x16x32_f16(
            false, afr[s].v, false, bf.v, (short)0, acc[s][t], false, false);
    }
  }

  // epilogue: lane holds col N=lm, rows r + 8*hi
  #pragma unroll
  for (int s = 0; s < 4; s++)
    #pragma unroll
    for (int t = 0; t < 4; t++)
      #pragma unroll
      for (int r = 0; r < 8; r++) {
        int m = gm + wm + s * 16 + r + 8 * hi;
        int n = gn + wn + t * 16 + lm;
        if (OUT_F32) ((float*)Cout)[(size_t)m * N + n] = acc[s][t][r];
        else ((_Float16*)Cout)[(size_t)m * N + n] = (_Float16)acc[s][t][r];
      }
}

// ---------------------------------------------------------------------------
// 4) RoPE + q scaling + rearrange: qkv fp16 [8192,1536] -> q,k,v fp16 [b*h,n,d]
// ---------------------------------------------------------------------------
__global__ __launch_bounds__(256) void rope_split_kernel(
    const _Float16* __restrict__ qkv, const float* __restrict__ rot,
    _Float16* __restrict__ q, _Float16* __restrict__ k,
    _Float16* __restrict__ v) {
  const int row = blockIdx.x;          // 0..8191
  const int b = row >> 12, n = row & (SEQ - 1);
  const _Float16* src = qkv + (size_t)row * DH3;
  for (int idx = threadIdx.x; idx < DH3; idx += 256) {
    int sel = idx >> 9;                // 0=q 1=k 2=v
    int h = (idx >> 6) & 7;
    int d = idx & 63;
    float val = (float)src[idx];
    if (sel < 2) {
      float ang = rot[n * DHEAD + d];
      // rotate_half: d<32 -> -x[d+32]; d>=32 -> x[d-32]
      float pv = (d < 32) ? -(float)src[idx + 32] : (float)src[idx - 32];
      val = val * cosf(ang) + pv * sinf(ang);
      if (sel == 0) val *= 0.125f;     // DHEAD^-0.5
    }
    size_t o = ((size_t)(b * HEADS + h) * SEQ + n) * DHEAD + d;
    if (sel == 0) q[o] = (_Float16)val;
    else if (sel == 1) k[o] = (_Float16)val;
    else v[o] = (_Float16)val;
  }
}

// ---------------------------------------------------------------------------
// 5) Flash attention (causal): grid (SEQ/64, b*h), 128 threads = 4 waves,
//    wave = 16 query rows. KV tiles of 64 staged in LDS (V via async-to-LDS,
//    K transposed through VGPRs). WMMA for QK^T and PV, online softmax.
// ---------------------------------------------------------------------------
__global__ __launch_bounds__(128) void flash_attn_kernel(
    const _Float16* __restrict__ q, const _Float16* __restrict__ k,
    const _Float16* __restrict__ v, _Float16* __restrict__ out) {
  __shared__ _Float16 sKT[64][72];     // K transposed: [d][key]
  __shared__ _Float16 sV[64][72];      // [key][d]
  __shared__ _Float16 sP[4][16][72];   // per-wave P scratch (D-layout -> A-layout)
  const int tid = threadIdx.x;
  const int lane = tid & 31, wid = tid >> 5;
  const int lm = lane & 15, hi = lane >> 4;
  const int bh = blockIdx.y;
  const int qbase = blockIdx.x * 64;
  const size_t base = (size_t)bh * SEQ * DHEAD;
  const int qrow0 = qbase + wid * 16;

  // Q fragments (A-layout), loaded once: rows qrow0+lm, d in two 32-chunks
  F16x16 qf[2];
  #pragma unroll
  for (int s = 0; s < 2; s++) {
    const _Float16* qp = q + base + (size_t)(qrow0 + lm) * DHEAD + s * 32;
    #pragma unroll
    for (int vv = 0; vv < 8; vv++) {
      int kk = ((vv < 4) ? 0 : 16) + hi * 8 + (vv & 3) * 2;
      qf[s].u[vv] = *(const unsigned int*)(qp + kk);
    }
  }

  float mstat[8], lstat[8];
  #pragma unroll
  for (int r = 0; r < 8; r++) { mstat[r] = -1e30f; lstat[r] = 0.f; }
  v8f oacc[4];
  #pragma unroll
  for (int t = 0; t < 4; t++) oacc[t] = (v8f)(0.f);

  const int nkv = blockIdx.x + 1;      // causal: only tiles <= query tile
  for (int it = 0; it < nkv; it++) {
    const int kv0 = it * 64;
    __syncthreads();
    // V tile via CDNA5 async global->LDS; K tile transposed through VGPRs
    #pragma unroll
    for (int i = 0; i < 4; i++) {
      int c = tid + i * 128;
      int r = c >> 3, c8 = (c & 7) * 8;
      async_copy_b128(&v[base + (size_t)(kv0 + r) * DHEAD + c8], &sV[r][c8]);
      v8h kl = *(const v8h*)&k[base + (size_t)(kv0 + r) * DHEAD + c8];
      #pragma unroll
      for (int e = 0; e < 8; e++) sKT[c8 + e][r] = kl[e];
    }
    asm volatile("s_wait_asynccnt 0" ::: "memory");
    __syncthreads();

    // S = Q * K^T : 4 key tiles of 16, accumulate over d (2 chunks of 32)
    v8f sacc[4];
    #pragma unroll
    for (int t = 0; t < 4; t++) sacc[t] = (v8f)(0.f);
    #pragma unroll
    for (int s = 0; s < 2; s++)
      #pragma unroll
      for (int t = 0; t < 4; t++) {
        F16x16 bf;
        #pragma unroll
        for (int e = 0; e < 16; e++)
          bf.h[e] = sKT[s * 32 + ((e < 8) ? 0 : 16) + hi * 8 + (e & 7)][t * 16 + lm];
        sacc[t] = __builtin_amdgcn_wmma_f32_16x16x32_f16(
            false, qf[s].v, false, bf.v, (short)0, sacc[t], false, false);
      }

    // causal mask + online softmax (row r+8*hi lives in one 16-lane group)
    float pv[4][8], rmax[8];
    #pragma unroll
    for (int r = 0; r < 8; r++) {
      int qrow = qrow0 + r + 8 * hi;
      float mx = -1e30f;
      #pragma unroll
      for (int t = 0; t < 4; t++) {
        int col = kv0 + t * 16 + lm;
        float sv = (col <= qrow) ? sacc[t][r] : -1e30f;
        pv[t][r] = sv;
        mx = fmaxf(mx, sv);
      }
      #pragma unroll
      for (int mk = 8; mk >= 1; mk >>= 1) mx = fmaxf(mx, __shfl_xor(mx, mk, 16));
      rmax[r] = mx;
    }
    #pragma unroll
    for (int r = 0; r < 8; r++) {
      float mn = fmaxf(mstat[r], rmax[r]);
      float alpha = __expf(mstat[r] - mn);
      mstat[r] = mn;
      float rs = 0.f;
      #pragma unroll
      for (int t = 0; t < 4; t++) {
        float p = __expf(pv[t][r] - mn);
        pv[t][r] = p;
        rs += p;
      }
      #pragma unroll
      for (int mk = 8; mk >= 1; mk >>= 1) rs += __shfl_xor(rs, mk, 16);
      lstat[r] = lstat[r] * alpha + rs;
      #pragma unroll
      for (int t = 0; t < 4; t++) oacc[t][r] *= alpha;
    }

    // write P (fp16) to per-wave LDS scratch, re-read in A-layout.
    #pragma unroll
    for (int t = 0; t < 4; t++)
      #pragma unroll
      for (int r = 0; r < 8; r++)
        sP[wid][r + 8 * hi][t * 16 + lm] = (_Float16)pv[t][r];
    // DS ops are in-order per wave; drain and fence the compiler.
    asm volatile("s_wait_dscnt 0" ::: "memory");

    // O += P * V
    #pragma unroll
    for (int s = 0; s < 2; s++) {
      F16x16 pf;
      #pragma unroll
      for (int vv = 0; vv < 8; vv++) {
        int kk = ((vv < 4) ? 0 : 16) + hi * 8 + (vv & 3) * 2;
        pf.u[vv] = *(const unsigned int*)&sP[wid][lm][s * 32 + kk];
      }
      #pragma unroll
      for (int t = 0; t < 4; t++) {
        F16x16 bf;
        #pragma unroll
        for (int e = 0; e < 16; e++)
          bf.h[e] = sV[s * 32 + ((e < 8) ? 0 : 16) + hi * 8 + (e & 7)][t * 16 + lm];
        oacc[t] = __builtin_amdgcn_wmma_f32_16x16x32_f16(
            false, pf.v, false, bf.v, (short)0, oacc[t], false, false);
      }
    }
  }

  // out: [b, n, h*d] fp16
  const int b = bh >> 3, h = bh & 7;
  #pragma unroll
  for (int t = 0; t < 4; t++)
    #pragma unroll
    for (int r = 0; r < 8; r++) {
      int qrow = qrow0 + r + 8 * hi;
      float o = oacc[t][r] / lstat[r];
      out[((size_t)(b * SEQ + qrow)) * DHID + h * DHEAD + t * 16 + lm] =
          (_Float16)o;
    }
}

// ---------------------------------------------------------------------------
// Launch
// ---------------------------------------------------------------------------
extern "C" void kernel_launch(void* const* d_in, const int* in_sizes, int n_in,
                              void* d_out, int out_size, void* d_ws, size_t ws_size,
                              hipStream_t stream) {
  (void)in_sizes; (void)n_in; (void)out_size; (void)ws_size;
  const float* x     = (const float*)d_in[0];
  const float* rot   = (const float*)d_in[1];
  const float* rmsw  = (const float*)d_in[2];
  const float* w_qkv = (const float*)d_in[3];
  const float* w_out = (const float*)d_in[4];
  float* outp = (float*)d_out;

  char* ws = (char*)d_ws;
  size_t off = 0;
  _Float16* xn    = (_Float16*)(ws + off); off += (size_t)ROWS * DIM * 2;        // 16 MB
  _Float16* wqh   = (_Float16*)(ws + off); off += (size_t)DIM * DH3 * 2;         // 3 MB
  _Float16* woh   = (_Float16*)(ws + off); off += (size_t)DHID * DIM * 2;        // 1 MB
  _Float16* qkv   = (_Float16*)(ws + off); off += (size_t)ROWS * DH3 * 2;        // 24 MB
  _Float16* qb    = (_Float16*)(ws + off); off += (size_t)B_ * HEADS * SEQ * DHEAD * 2;
  _Float16* kb    = (_Float16*)(ws + off); off += (size_t)B_ * HEADS * SEQ * DHEAD * 2;
  _Float16* vb    = (_Float16*)(ws + off); off += (size_t)B_ * HEADS * SEQ * DHEAD * 2;
  _Float16* aout  = (_Float16*)(ws + off); off += (size_t)ROWS * DHID * 2;       // 8 MB

  rmsnorm_kernel<<<ROWS, 256, 0, stream>>>(x, rmsw, xn);

  int nwq = DIM * DH3;
  f32_to_f16_kernel<<<(nwq + 255) / 256, 256, 0, stream>>>(w_qkv, wqh, nwq);
  int nwo = DHID * DIM;
  f32_to_f16_kernel<<<(nwo + 255) / 256, 256, 0, stream>>>(w_out, woh, nwo);

  gemm_f16_kernel<false><<<dim3(DH3 / 256, ROWS / 128), 256, 0, stream>>>(
      xn, wqh, (void*)qkv, ROWS, DH3, DIM);

  rope_split_kernel<<<ROWS, 256, 0, stream>>>(qkv, rot, qb, kb, vb);

  flash_attn_kernel<<<dim3(SEQ / 64, B_ * HEADS), 128, 0, stream>>>(
      qb, kb, vb, aout);

  gemm_f16_kernel<true><<<dim3(DIM / 256, ROWS / 128), 256, 0, stream>>>(
      aout, woh, (void*)outp, ROWS, DIM, DHID);
}
